// AttentionGuidedEmbedding_77438260347445
// MI455X (gfx1250) — matrix operand order repaired and checked by reading, not compiled, not run on hardware
//
#include <hip/hip_runtime.h>
#include <hip/hip_bf16.h>

typedef __attribute__((ext_vector_type(16))) _Float16 v16h;
typedef __attribute__((ext_vector_type(8)))  _Float16 v8h;
typedef __attribute__((ext_vector_type(8)))  float    v8f;
typedef __attribute__((ext_vector_type(4)))  float    v4f;

#define VOCAB   50257
#define EDIM    128
#define NDOM    16
#define DSZ     256
#define NTOK    32768      // B*S = 16*2048
#define TILE    16         // tokens per wave
#define WPB     2          // waves per block
#define CORRSC  0.1f

#define NFRAG1  (NDOM * DSZ * EDIM)   // 524288 f16 elems for W1 frags
#define NFRAG2  (NDOM * EDIM * DSZ)   // 524288 f16 elems for W2 frags
#define DOMH    (DSZ * EDIM)          // 32768 halves (64KB) per domain per array

// ---------------------------------------------------------------------------
// One-shot: convert W1/W2 (f32) into f16 B-fragments pre-swizzled for the
// 16x16x32 WMMA B layout: [d][ntile][kchunk][lane][16 contiguous halves].
// ---------------------------------------------------------------------------
__global__ void convert_weights_kernel(const float* __restrict__ W1,
                                       const float* __restrict__ W2,
                                       _Float16* __restrict__ wf1,
                                       _Float16* __restrict__ wf2)
{
    const int i = blockIdx.x * blockDim.x + threadIdx.x;
    if (i < NFRAG1) {
        // layout: (((d*16 + nt)*4 + kc)*32 + lane)*16 + e
        const int e    = i & 15;
        const int lane = (i >> 4) & 31;
        const int kc   = (i >> 9) & 3;
        const int nt   = (i >> 11) & 15;
        const int d    = (i >> 15) & 15;
        const size_t src = ((size_t)(d * DSZ + nt * 16 + (lane & 15))) * EDIM
                         + kc * 32 + (lane >> 4) * 16 + e;
        wf1[i] = (_Float16)W1[src];
    } else if (i < NFRAG1 + NFRAG2) {
        const int j    = i - NFRAG1;
        // layout: (((d*8 + nt2)*8 + kc2)*32 + lane)*16 + e
        const int e    = j & 15;
        const int lane = (j >> 4) & 31;
        const int kc2  = (j >> 9) & 7;
        const int nt2  = (j >> 12) & 7;
        const int d    = (j >> 15) & 15;
        const size_t src = ((size_t)(d * EDIM + nt2 * 16 + (lane & 15))) * DSZ
                         + kc2 * 32 + (lane >> 4) * 16 + e;
        wf2[j] = (_Float16)W2[src];
    }
}

// ---------------------------------------------------------------------------
// Main kernel: one wave owns a 16-token tile and runs the whole 16-domain
// sequential scan with h carried in f32 LDS.
// ---------------------------------------------------------------------------
template<bool PRE>
__global__ __launch_bounds__(WPB * 32)
void domain_scan_kernel(const int*   __restrict__ x,
                        const float* __restrict__ base_embed,
                        const float* __restrict__ W1,       // [NDOM][DSZ][EDIM]
                        const float* __restrict__ W2,       // [NDOM][EDIM][DSZ]
                        const int*   __restrict__ membership,
                        const _Float16* __restrict__ wf1,   // pre-swizzled frags
                        const _Float16* __restrict__ wf2,
                        float*       __restrict__ out)      // [NTOK][EDIM]
{
    __shared__ __align__(32) float    hbuf[WPB][TILE][EDIM];
    __shared__ __align__(32) _Float16 midbuf[WPB][TILE][DSZ];
    __shared__ int   tokbuf[WPB][TILE];
    __shared__ float maskbuf[WPB][TILE];

    const int tid  = threadIdx.x;
    const int wave = tid >> 5;
    const int lane = tid & 31;
    const int tile = blockIdx.x * WPB + wave;
    if (tile >= NTOK / TILE) return;                 // uniform per wave

    const int mrow  = lane & 15;
    const int hi    = lane >> 4;
    const int kbase = hi * 16;                       // A-layout K base (0 or 16)

    // ---- tokens + initial embeddings ----
    if (lane < TILE) tokbuf[wave][lane] = x[tile * TILE + lane];
    __syncthreads();

    for (int m = 0; m < TILE; ++m) {
        const int t = tokbuf[wave][m];
        *(v4f*)&hbuf[wave][m][lane * 4] =
            *(const v4f*)(base_embed + (size_t)t * EDIM + lane * 4);
    }
    __syncthreads();

    // ---- sequential scan over domains ----
    #pragma unroll 1
    for (int d = 0; d < NDOM; ++d) {
        if (lane < TILE) {
            const int mb = membership[(size_t)d * VOCAB + tokbuf[wave][lane]];
            maskbuf[wave][lane] = (mb != 0) ? CORRSC : 0.0f;
        }
        __syncthreads();

        // Speculative prefetch of next domain's fragment regions (WGP$/L2 warm).
        if constexpr (PRE) {
            if (d + 1 < NDOM) {
                const char* p1 = (const char*)(wf1 + (size_t)(d + 1) * DOMH) + lane * 128;
                const char* p2 = (const char*)(wf2 + (size_t)(d + 1) * DOMH) + lane * 128;
                #pragma unroll
                for (int i = 0; i < 16; ++i) {
                    __builtin_prefetch(p1 + i * 4096, 0, 1);
                    __builtin_prefetch(p2 + i * 4096, 0, 1);
                }
            }
        }

        // A-frags for GEMM1 from f32 h (16-bit A 16x32 layout), vector LDS loads.
        v16h a1[4];
        #pragma unroll
        for (int kc = 0; kc < 4; ++kc) {
            const float* hrow = &hbuf[wave][mrow][kc * 32 + kbase];
            const v4f f0 = *(const v4f*)(hrow);
            const v4f f1 = *(const v4f*)(hrow + 4);
            const v4f f2 = *(const v4f*)(hrow + 8);
            const v4f f3 = *(const v4f*)(hrow + 12);
            v16h a;
            #pragma unroll
            for (int e = 0; e < 4; ++e) {
                a[e]      = (_Float16)f0[e];
                a[e + 4]  = (_Float16)f1[e];
                a[e + 8]  = (_Float16)f2[e];
                a[e + 12] = (_Float16)f3[e];
            }
            a1[kc] = a;
        }

        // frag index: ((d*16 + nt)*4 + kc)*32 + lane = d*2048 + nt*128 + kc*32 + lane
        const v16h* bp1 = (const v16h*)wf1 + (size_t)d * 2048 + lane;
        const v16h* bp2 = (const v16h*)wf2 + (size_t)d * 2048 + lane;

        auto load_b1 = [&](v16h (&bf)[4], int nt) {
            if constexpr (PRE) {
                #pragma unroll
                for (int kc = 0; kc < 4; ++kc) bf[kc] = bp1[nt * 128 + kc * 32];
            } else {
                #pragma unroll
                for (int kc = 0; kc < 4; ++kc) {
                    const float* w1p = W1 + (((size_t)d * DSZ + nt * 16 + mrow) * EDIM)
                                          + kc * 32 + hi * 16;
                    #pragma unroll
                    for (int e = 0; e < 16; ++e) bf[kc][e] = (_Float16)w1p[e];
                }
            }
        };
        auto tile1 = [&](const v16h (&bf)[4]) -> v8f {
            v8f acc = {};
            #pragma unroll
            for (int kc = 0; kc < 4; ++kc)
                acc = __builtin_amdgcn_wmma_f32_16x16x32_f16(
                        false, a1[kc], false, bf[kc], (short)0, acc, false, false);
            return acc;
        };
        auto gelu_store = [&](const v8f& acc, int nt) {
            #pragma unroll
            for (int r = 0; r < 8; ++r) {
                const float v = acc[r];
                const float g = 0.5f * v * (1.0f + erff(v * 0.70710678f));
                midbuf[wave][r + hi * 8][nt * 16 + mrow] = (_Float16)g;
            }
        };

        // ---- GEMM1: mid = gelu(h @ W1[d]^T), double-buffered B-frags ----
        {
            v16h bA[4], bB[4];
            load_b1(bA, 0);
            #pragma unroll 1
            for (int ntp = 0; ntp < 8; ++ntp) {
                v8f acc = tile1(bA);
                load_b1(bB, 2 * ntp + 1);        // in flight during gelu
                gelu_store(acc, 2 * ntp);
                acc = tile1(bB);
                if (ntp < 7) load_b1(bA, 2 * ntp + 2);
                gelu_store(acc, 2 * ntp + 1);
            }
        }
        __syncthreads();

        float ms[8];
        #pragma unroll
        for (int r = 0; r < 8; ++r) ms[r] = maskbuf[wave][r + hi * 8];

        // ---- GEMM2: h += scale * (mid @ W2[d]^T) ----
        #pragma unroll 1
        for (int nt2 = 0; nt2 < EDIM / 16; ++nt2) {
            // batch all 8 K-chunk B-frags -> one clause, incremental waits
            v16h b2[8];
            if constexpr (PRE) {
                #pragma unroll
                for (int kc2 = 0; kc2 < 8; ++kc2)
                    b2[kc2] = bp2[nt2 * 256 + kc2 * 32];
            } else {
                #pragma unroll
                for (int kc2 = 0; kc2 < 8; ++kc2) {
                    const float* w2p = W2 + (((size_t)d * EDIM + nt2 * 16 + mrow) * DSZ)
                                          + kc2 * 32 + hi * 16;
                    #pragma unroll
                    for (int e = 0; e < 16; ++e) b2[kc2][e] = (_Float16)w2p[e];
                }
            }
            v8f acc2 = {};
            #pragma unroll
            for (int kc2 = 0; kc2 < 8; ++kc2) {
                const _Float16* mp = &midbuf[wave][mrow][kc2 * 32 + kbase];
                const v8h g0 = *(const v8h*)(mp);
                const v8h g1 = *(const v8h*)(mp + 8);
                const v16h a2 = __builtin_shufflevector(g0, g1,
                        0,1,2,3,4,5,6,7,8,9,10,11,12,13,14,15);
                acc2 = __builtin_amdgcn_wmma_f32_16x16x32_f16(
                        false, a2, false, b2[kc2], (short)0, acc2, false, false);
            }
            #pragma unroll
            for (int r = 0; r < 8; ++r)
                hbuf[wave][r + hi * 8][nt2 * 16 + mrow] += ms[r] * acc2[r];
        }
        __syncthreads();
    }

    // ---- write final h (f32) ----
    for (int m = 0; m < TILE; ++m) {
        float* dst = out + ((size_t)tile * TILE + m) * EDIM;
        *(v4f*)(dst + lane * 4) = *(const v4f*)&hbuf[wave][m][lane * 4];
    }
}

extern "C" void kernel_launch(void* const* d_in, const int* in_sizes, int n_in,
                              void* d_out, int out_size, void* d_ws, size_t ws_size,
                              hipStream_t stream) {
    (void)in_sizes; (void)n_in; (void)out_size;
    const int*   x          = (const int*)  d_in[0];
    const float* base_embed = (const float*)d_in[1];
    const float* W1         = (const float*)d_in[2];
    const float* W2         = (const float*)d_in[3];
    const int*   membership = (const int*)  d_in[4];
    float*       out        = (float*)d_out;

    const int ntiles = NTOK / TILE;          // 2048
    const int blocks = ntiles / WPB;         // 1024

    const size_t need = (size_t)(NFRAG1 + NFRAG2) * sizeof(_Float16); // 2 MB
    if (ws_size >= need) {
        _Float16* wf1 = (_Float16*)d_ws;
        _Float16* wf2 = wf1 + NFRAG1;
        const int ntot = NFRAG1 + NFRAG2;
        convert_weights_kernel<<<(ntot + 255) / 256, 256, 0, stream>>>(W1, W2, wf1, wf2);
        domain_scan_kernel<true><<<blocks, WPB * 32, 0, stream>>>(
            x, base_embed, W1, W2, membership, wf1, wf2, out);
    } else {
        domain_scan_kernel<false><<<blocks, WPB * 32, 0, stream>>>(
            x, base_embed, W1, W2, membership, nullptr, nullptr, out);
    }
}
